// BailingMoe_55860344652005
// MI455X (gfx1250) — matrix operand order repaired and checked by reading, not compiled, not run on hardware
//
#include <hip/hip_runtime.h>

// ---------------------------------------------------------------------------
// BailingMoE for MI455X (gfx1250, wave32, WMMA).
// fp32 in/out. x and intermediate activations kept as bf16 (x converted once;
// gate_up epilogue emits bf16), weights converted fp32->bf16 during LDS
// staging. All GEMMs: v_wmma_f32_16x16x32_bf16, fp32 accumulation, 64x128
// block tiles, double-buffered LDS so global loads + conversion overlap WMMA.
// Sparse routing (top-4 of 16) via per-expert gather lists: grouped GEMMs do
// only K/E = 1/4 of the dense reference work.
// ---------------------------------------------------------------------------

#define T_TOK  1024
#define HDIM   1024
#define NEXP   16
#define KTOP   4
#define IDIM   1408
#define ISDIM  2816

typedef __attribute__((ext_vector_type(16))) __bf16          v16bf;
typedef __attribute__((ext_vector_type(8)))  __bf16          v8bf;
typedef __attribute__((ext_vector_type(8)))  float           v8f;
typedef __attribute__((ext_vector_type(8)))  unsigned short  us8;

__device__ __forceinline__ unsigned short f2bf(float f) {
  unsigned u = __builtin_bit_cast(unsigned, f);
  unsigned r = u + 0x7FFFu + ((u >> 16) & 1u);   // round-to-nearest-even
  return (unsigned short)(r >> 16);
}

// Pack two fp32 -> packed bf16x2 (low = a, high = b).
__device__ __forceinline__ unsigned pack2(float a, float b) {
#if __has_builtin(__builtin_amdgcn_cvt_pk_bf16_f32)
  return __builtin_bit_cast(unsigned, __builtin_amdgcn_cvt_pk_bf16_f32(a, b));
#else
  unsigned ua = __builtin_bit_cast(unsigned, a);
  unsigned ub = __builtin_bit_cast(unsigned, b);
  ua += 0x7FFFu + ((ua >> 16) & 1u);
  ub += 0x7FFFu + ((ub >> 16) & 1u);
  // one v_perm_b32: result bytes {ub[3],ub[2],ua[3],ua[2]}
  return __builtin_amdgcn_perm(ub, ua, 0x07060302u);
#endif
}

// ---------------------------------------------------------------------------
// x fp32 -> bf16 copy (done once; x is re-read ~66x by the GEMMs).
// ---------------------------------------------------------------------------
__global__ __launch_bounds__(256)
void cvt_bf16_kernel(const float* __restrict__ src,
                     unsigned short* __restrict__ dst, int n) {
  int i = (blockIdx.x * 256 + threadIdx.x) * 4;
  if (i < n) {
    float4 f = *(const float4*)(src + i);
    uint2 v;
    v.x = pack2(f.x, f.y);
    v.y = pack2(f.z, f.w);
    *(uint2*)(dst + i) = v;
  }
}

// ---------------------------------------------------------------------------
// Router: logits -> softmax top-4 renorm -> per-expert (token, weight) lists.
// ---------------------------------------------------------------------------
__global__ __launch_bounds__(256)
void router_kernel(const float* __restrict__ x, const float* __restrict__ wg,
                   int* __restrict__ cnt, int* __restrict__ tok,
                   float* __restrict__ wts) {
  const int t   = blockIdx.x;
  const int tid = threadIdx.x;
  __shared__ float sl[NEXP];
  const int e = tid >> 4;
  const int l = tid & 15;
  float p = 0.f;
  for (int h = l; h < HDIM; h += 16)
    p += x[(size_t)t * HDIM + h] * wg[(size_t)h * NEXP + e];
  for (int off = 8; off > 0; off >>= 1)
    p += __shfl_xor(p, off, 16);
  if (l == 0) sl[e] = p;
  __syncthreads();
  if (tid == 0) {
    float mx = sl[0];
    for (int i = 1; i < NEXP; ++i) mx = fmaxf(mx, sl[i]);
    float pr[NEXP];
    for (int i = 0; i < NEXP; ++i) pr[i] = __expf(sl[i] - mx);
    bool taken[NEXP] = {};
    int ids[KTOP]; float pw[KTOP]; float s = 0.f;
    for (int k = 0; k < KTOP; ++k) {
      int best = 0; float bv = -1.f;
      for (int i = 0; i < NEXP; ++i)
        if (!taken[i] && pr[i] > bv) { bv = pr[i]; best = i; }
      taken[best] = true; ids[k] = best; pw[k] = bv; s += bv;
    }
    float inv = 1.f / s;
    for (int k = 0; k < KTOP; ++k) {
      int pos = atomicAdd(&cnt[ids[k]], 1);
      tok[ids[k] * T_TOK + pos] = t;
      wts[ids[k] * T_TOK + pos] = pw[k] * inv;
    }
  }
}

__global__ void scan_kernel(const int* __restrict__ cnt, int* __restrict__ offs) {
  if (threadIdx.x == 0) {
    int a = 0;
    for (int i = 0; i < NEXP; ++i) { offs[i] = a; a += cnt[i]; }
  }
}

// ---------------------------------------------------------------------------
// Fused gate_up GEMM + SiLU*mul, bf16 A, fp32 B (converted while staging),
// bf16 act output. Double-buffered LDS.
// ---------------------------------------------------------------------------
__global__ __launch_bounds__(256)
void gateup_kernel(const unsigned short* __restrict__ A,   // bf16 [.,lda]
                   const float* __restrict__ B,            // fp32 [K,ldb]
                   unsigned short* __restrict__ Cact,      // bf16 [.,ldc]
                   const int* __restrict__ tok, const int* __restrict__ cnt,
                   const int* __restrict__ offs,
                   int Mmax, int Kdim, int lda, int ldb, int ldc, int uoff,
                   long bstride) {
  const int e   = blockIdx.z;
  const int m0  = blockIdx.y * 64;
  const int bn0 = blockIdx.x * 128;
  const int mrows = cnt ? cnt[e] : Mmax;
  if (m0 >= mrows) return;
  const int crow0 = offs ? offs[e] : 0;
  const float* Bg = B + (size_t)e * bstride;
  const float* Bu = Bg + uoff;

  __shared__ alignas(64) unsigned short As [2][64][32];
  __shared__ alignas(64) unsigned short Bsg[2][128][32];   // transposed [n][k]
  __shared__ alignas(64) unsigned short Bsu[2][128][32];

  const int tid  = threadIdx.x;
  const int lane = tid & 31;
  const int wave = tid >> 5;
  const int mt   = (wave >> 1) * 16;
  const int nh   = (wave & 1) * 64;

  v8f accg[4], accu[4];
  const v8f vzero = {0.f,0.f,0.f,0.f,0.f,0.f,0.f,0.f};
#pragma unroll
  for (int c = 0; c < 4; ++c) { accg[c] = vzero; accu[c] = vzero; }

  // A staging: 4 threads x 8 bf16 per row (one b128 each)
  const int ar  = tid >> 2;
  const int akq = (tid & 3) * 8;
  int garow = -1;
  {
    int lrow = m0 + ar;
    if (lrow < mrows) garow = tok ? tok[e * T_TOK + lrow] : lrow;
  }
  const us8 z8 = {0,0,0,0,0,0,0,0};

  // ---- prologue: stage K-step 0 into buffer 0 ----
  {
    us8 av = z8;
    if (garow >= 0) av = *(const us8*)(A + (size_t)garow * lda + akq);
    *(us8*)&As[0][ar][akq] = av;
#pragma unroll
    for (int j = 0; j < 2; ++j) {
      int it = tid + j * 256;
      int kp = (it & 15) * 2;
      int nq = (it >> 4) * 4;
      const float* pbg = Bg + (size_t)kp * ldb + bn0 + nq;
      float4 g0 = *(const float4*)pbg;
      float4 g1 = *(const float4*)(pbg + ldb);
      const float* pbu = Bu + (size_t)kp * ldb + bn0 + nq;
      float4 u0 = *(const float4*)pbu;
      float4 u1 = *(const float4*)(pbu + ldb);
      *(unsigned*)&Bsg[0][nq+0][kp] = pack2(g0.x, g1.x);
      *(unsigned*)&Bsg[0][nq+1][kp] = pack2(g0.y, g1.y);
      *(unsigned*)&Bsg[0][nq+2][kp] = pack2(g0.z, g1.z);
      *(unsigned*)&Bsg[0][nq+3][kp] = pack2(g0.w, g1.w);
      *(unsigned*)&Bsu[0][nq+0][kp] = pack2(u0.x, u1.x);
      *(unsigned*)&Bsu[0][nq+1][kp] = pack2(u0.y, u1.y);
      *(unsigned*)&Bsu[0][nq+2][kp] = pack2(u0.z, u1.z);
      *(unsigned*)&Bsu[0][nq+3][kp] = pack2(u0.w, u1.w);
    }
  }

  const int nsteps = Kdim / 32;
  int buf = 0;
  for (int s = 0; s < nsteps; ++s) {
    __syncthreads();
    const bool more = (s + 1) < nsteps;

    // ---- issue next-tile global loads (hidden behind WMMA below) ----
    us8 av = z8;
    float4 g0[2] = {}, g1[2] = {}, u0[2] = {}, u1[2] = {};
    if (more) {
      const int k0 = (s + 1) * 32;
      if (garow >= 0) av = *(const us8*)(A + (size_t)garow * lda + k0 + akq);
#pragma unroll
      for (int j = 0; j < 2; ++j) {
        int it = tid + j * 256;
        int kp = (it & 15) * 2;
        int nq = (it >> 4) * 4;
        const float* pbg = Bg + (size_t)(k0 + kp) * ldb + bn0 + nq;
        g0[j] = *(const float4*)pbg;
        g1[j] = *(const float4*)(pbg + ldb);
        const float* pbu = Bu + (size_t)(k0 + kp) * ldb + bn0 + nq;
        u0[j] = *(const float4*)pbu;
        u1[j] = *(const float4*)(pbu + ldb);
      }
    }

    // ---- WMMA on current buffer ----
    {
      const int arw = mt + (lane & 15);
      const int kh8 = (lane >> 4) * 16;
      v8bf alo = *(const v8bf*)&As[buf][arw][kh8];
      v8bf ahi = *(const v8bf*)&As[buf][arw][8 + kh8];
      v16bf af = __builtin_shufflevector(alo, ahi,
                   0,1,2,3,4,5,6,7,8,9,10,11,12,13,14,15);
      const int bn = nh + (lane & 15);
      const int kb = (lane >> 4) * 16;
#pragma unroll
      for (int c = 0; c < 4; ++c) {
        v16bf bg = *(const v16bf*)&Bsg[buf][bn + c*16][kb];
        accg[c] = __builtin_amdgcn_wmma_f32_16x16x32_bf16(
                    false, af, false, bg, (short)0, accg[c], false, false);
        v16bf bu = *(const v16bf*)&Bsu[buf][bn + c*16][kb];
        accu[c] = __builtin_amdgcn_wmma_f32_16x16x32_bf16(
                    false, af, false, bu, (short)0, accu[c], false, false);
      }
    }

    // ---- convert + store next tile into other buffer ----
    if (more) {
      const int nb = buf ^ 1;
      *(us8*)&As[nb][ar][akq] = av;
#pragma unroll
      for (int j = 0; j < 2; ++j) {
        int it = tid + j * 256;
        int kp = (it & 15) * 2;
        int nq = (it >> 4) * 4;
        *(unsigned*)&Bsg[nb][nq+0][kp] = pack2(g0[j].x, g1[j].x);
        *(unsigned*)&Bsg[nb][nq+1][kp] = pack2(g0[j].y, g1[j].y);
        *(unsigned*)&Bsg[nb][nq+2][kp] = pack2(g0[j].z, g1[j].z);
        *(unsigned*)&Bsg[nb][nq+3][kp] = pack2(g0[j].w, g1[j].w);
        *(unsigned*)&Bsu[nb][nq+0][kp] = pack2(u0[j].x, u1[j].x);
        *(unsigned*)&Bsu[nb][nq+1][kp] = pack2(u0[j].y, u1[j].y);
        *(unsigned*)&Bsu[nb][nq+2][kp] = pack2(u0[j].z, u1[j].z);
        *(unsigned*)&Bsu[nb][nq+3][kp] = pack2(u0[j].w, u1[j].w);
      }
    }
    buf ^= 1;
  }

  // ---- epilogue: act = silu(g) * u -> bf16, compact rows ----
  const int rbase = m0 + mt + ((lane >> 4) ? 8 : 0);
  const int colb  = bn0 + nh + (lane & 15);
#pragma unroll
  for (int c = 0; c < 4; ++c) {
#pragma unroll
    for (int r = 0; r < 8; ++r) {
      int grow = rbase + r;
      if (grow < mrows) {
        float g = accg[c][r], u = accu[c][r];
        float sv = g / (1.f + __expf(-g));
        Cact[(size_t)(crow0 + grow) * ldc + colb + c*16] = f2bf(sv * u);
      }
    }
  }
}

// ---------------------------------------------------------------------------
// Down-projection GEMM, bf16 A (activations), fp32 B. Double-buffered LDS.
// Shared path: plain store. Routed path: scaled scatter-add via fp32 atomics.
// ---------------------------------------------------------------------------
__global__ __launch_bounds__(256)
void down_kernel(const unsigned short* __restrict__ A,  // bf16 [.,lda]
                 const float* __restrict__ B,
                 float* __restrict__ C,
                 const int* __restrict__ tok, const float* __restrict__ wts,
                 const int* __restrict__ cnt, const int* __restrict__ offs,
                 int Mmax, int Kdim, int lda, int ldb, int ldc, long bstride) {
  const int e   = blockIdx.z;
  const int m0  = blockIdx.y * 64;
  const int bn0 = blockIdx.x * 128;
  const int mrows = cnt ? cnt[e] : Mmax;
  if (m0 >= mrows) return;
  const int arow0 = offs ? offs[e] : 0;
  const float* Be = B + (size_t)e * bstride;

  __shared__ alignas(64) unsigned short As[2][64][32];
  __shared__ alignas(64) unsigned short Bs[2][128][32];

  const int tid  = threadIdx.x;
  const int lane = tid & 31;
  const int wave = tid >> 5;
  const int mt   = (wave >> 1) * 16;
  const int nh   = (wave & 1) * 64;

  v8f acc[4];
  const v8f vzero = {0.f,0.f,0.f,0.f,0.f,0.f,0.f,0.f};
#pragma unroll
  for (int c = 0; c < 4; ++c) acc[c] = vzero;

  const int ar  = tid >> 2;
  const int akq = (tid & 3) * 8;
  int garow = -1;
  {
    int lrow = m0 + ar;
    if (lrow < mrows) garow = arow0 + lrow;
  }
  const us8 z8 = {0,0,0,0,0,0,0,0};

  // prologue: stage step 0
  {
    us8 av = z8;
    if (garow >= 0) av = *(const us8*)(A + (size_t)garow * lda + akq);
    *(us8*)&As[0][ar][akq] = av;
#pragma unroll
    for (int j = 0; j < 2; ++j) {
      int it = tid + j * 256;
      int kp = (it & 15) * 2;
      int nq = (it >> 4) * 4;
      const float* pb = Be + (size_t)kp * ldb + bn0 + nq;
      float4 b0 = *(const float4*)pb;
      float4 b1 = *(const float4*)(pb + ldb);
      *(unsigned*)&Bs[0][nq+0][kp] = pack2(b0.x, b1.x);
      *(unsigned*)&Bs[0][nq+1][kp] = pack2(b0.y, b1.y);
      *(unsigned*)&Bs[0][nq+2][kp] = pack2(b0.z, b1.z);
      *(unsigned*)&Bs[0][nq+3][kp] = pack2(b0.w, b1.w);
    }
  }

  const int nsteps = Kdim / 32;
  int buf = 0;
  for (int s = 0; s < nsteps; ++s) {
    __syncthreads();
    const bool more = (s + 1) < nsteps;

    us8 av = z8;
    float4 b0[2] = {}, b1[2] = {};
    if (more) {
      const int k0 = (s + 1) * 32;
      if (garow >= 0) av = *(const us8*)(A + (size_t)garow * lda + k0 + akq);
#pragma unroll
      for (int j = 0; j < 2; ++j) {
        int it = tid + j * 256;
        int kp = (it & 15) * 2;
        int nq = (it >> 4) * 4;
        const float* pb = Be + (size_t)(k0 + kp) * ldb + bn0 + nq;
        b0[j] = *(const float4*)pb;
        b1[j] = *(const float4*)(pb + ldb);
      }
    }

    {
      const int arw = mt + (lane & 15);
      const int kh8 = (lane >> 4) * 16;
      v8bf alo = *(const v8bf*)&As[buf][arw][kh8];
      v8bf ahi = *(const v8bf*)&As[buf][arw][8 + kh8];
      v16bf af = __builtin_shufflevector(alo, ahi,
                   0,1,2,3,4,5,6,7,8,9,10,11,12,13,14,15);
      const int bn = nh + (lane & 15);
      const int kb = (lane >> 4) * 16;
#pragma unroll
      for (int c = 0; c < 4; ++c) {
        v16bf bf = *(const v16bf*)&Bs[buf][bn + c*16][kb];
        acc[c] = __builtin_amdgcn_wmma_f32_16x16x32_bf16(
                   false, af, false, bf, (short)0, acc[c], false, false);
      }
    }

    if (more) {
      const int nb = buf ^ 1;
      *(us8*)&As[nb][ar][akq] = av;
#pragma unroll
      for (int j = 0; j < 2; ++j) {
        int it = tid + j * 256;
        int kp = (it & 15) * 2;
        int nq = (it >> 4) * 4;
        *(unsigned*)&Bs[nb][nq+0][kp] = pack2(b0[j].x, b1[j].x);
        *(unsigned*)&Bs[nb][nq+1][kp] = pack2(b0[j].y, b1[j].y);
        *(unsigned*)&Bs[nb][nq+2][kp] = pack2(b0[j].z, b1[j].z);
        *(unsigned*)&Bs[nb][nq+3][kp] = pack2(b0[j].w, b1[j].w);
      }
    }
    buf ^= 1;
  }

  const int rbase = mt + ((lane >> 4) ? 8 : 0);
  const int colb  = bn0 + nh + (lane & 15);
#pragma unroll
  for (int c = 0; c < 4; ++c) {
#pragma unroll
    for (int r = 0; r < 8; ++r) {
      int lrow = m0 + rbase + r;
      if (lrow < mrows) {
        float v = acc[c][r];
        int col = colb + c * 16;
        if (tok) {
          int trow = tok[e * T_TOK + lrow];
          atomicAdd(&C[(size_t)trow * ldc + col], wts[e * T_TOK + lrow] * v);
        } else {
          C[(size_t)lrow * ldc + col] = v;
        }
      }
    }
  }
}

// ---------------------------------------------------------------------------
extern "C" void kernel_launch(void* const* d_in, const int* in_sizes, int n_in,
                              void* d_out, int out_size, void* d_ws, size_t ws_size,
                              hipStream_t stream) {
  (void)in_sizes; (void)n_in; (void)out_size; (void)ws_size;
  const float* x    = (const float*)d_in[0];
  const float* wg   = (const float*)d_in[1];
  const float* wmgu = (const float*)d_in[2];
  const float* wmd  = (const float*)d_in[3];
  const float* wsgu = (const float*)d_in[4];
  const float* wsd  = (const float*)d_in[5];
  float* out = (float*)d_out;

  char* ws = (char*)d_ws;
  size_t o = 0;
  int*   cnt   = (int*)(ws + o);  o += 256;
  int*   offs  = (int*)(ws + o);  o += 256;
  int*   tok   = (int*)(ws + o);  o += (size_t)NEXP * T_TOK * 4;
  float* wts   = (float*)(ws + o); o += (size_t)NEXP * T_TOK * 4;
  unsigned short* xb    = (unsigned short*)(ws + o); o += (size_t)T_TOK * HDIM * 2;          // 2 MB
  unsigned short* act_s = (unsigned short*)(ws + o); o += (size_t)T_TOK * ISDIM * 2;         // 5.8 MB
  unsigned short* act_e = (unsigned short*)(ws + o); o += (size_t)T_TOK * KTOP * IDIM * 2;   // 11.5 MB

  hipMemsetAsync(cnt, 0, NEXP * sizeof(int), stream);
  cvt_bf16_kernel<<<(T_TOK * HDIM) / 1024, 256, 0, stream>>>(x, xb, T_TOK * HDIM);
  router_kernel<<<T_TOK, 256, 0, stream>>>(x, wg, cnt, tok, wts);
  scan_kernel<<<1, 32, 0, stream>>>(cnt, offs);

  // shared experts: gate_up + silu*mul  (M=1024, Nhalf=2816, K=1024)
  gateup_kernel<<<dim3(ISDIM/128, T_TOK/64, 1), 256, 0, stream>>>(
      xb, wsgu, act_s, nullptr, nullptr, nullptr,
      T_TOK, HDIM, HDIM, 2*ISDIM, ISDIM, ISDIM, 0L);

  // routed experts: grouped gate_up over gathered token rows
  gateup_kernel<<<dim3(IDIM/128, T_TOK/64, NEXP), 256, 0, stream>>>(
      xb, wmgu, act_e, tok, cnt, offs,
      T_TOK, HDIM, HDIM, 2*IDIM, IDIM, IDIM, (long)HDIM * 2 * IDIM);

  // shared down: writes out (full coverage) before routed adds
  down_kernel<<<dim3(HDIM/128, T_TOK/64, 1), 256, 0, stream>>>(
      act_s, wsd, out, nullptr, nullptr, nullptr, nullptr,
      T_TOK, ISDIM, ISDIM, HDIM, HDIM, 0L);

  // routed down: scaled scatter-add into out
  down_kernel<<<dim3(HDIM/128, T_TOK/64, NEXP), 256, 0, stream>>>(
      act_e, wmd, out, tok, wts, cnt, offs,
      T_TOK, IDIM, IDIM, HDIM, HDIM, (long)IDIM * HDIM);
}
